// AutoCorrelation_34402688041210
// MI455X (gfx1250) — matrix operand order repaired and checked by reading, not compiled, not run on hardware
//
#include <hip/hip_runtime.h>
#include <hip/hip_bf16.h>

typedef __attribute__((ext_vector_type(16))) __bf16 v16bf;
typedef __attribute__((ext_vector_type(8)))  float  v8f;

#define T_LEN 2016
#define NSEQ  2048   // B*N*H = 2*128*8
#define NBN   256    // B*N

// ---------------- Kernel 1: projection q/k = Concat(Q,t) @ W^T ----------------
__global__ __launch_bounds__(256) void proj_kernel(
    const float* __restrict__ Q, const float* __restrict__ K,
    const float* __restrict__ tt, const float* __restrict__ W,
    float* __restrict__ q_ws, float* __restrict__ k_ws)
{
  __shared__ float Ws[8 * 65];
  for (int i = threadIdx.x; i < 520; i += 256) Ws[i] = W[i];
  __syncthreads();

  int bn = blockIdx.x >> 3;
  int t  = ((blockIdx.x & 7) << 8) + threadIdx.x;
  if (t >= T_LEN) return;
  int b = bn >> 7;
  float tval = tt[b * T_LEN + t];
  size_t row = ((size_t)bn * T_LEN + t) * 64;

  { // Q pass
    float r[64];
    const float4* p = (const float4*)(Q + row);
#pragma unroll
    for (int f4 = 0; f4 < 16; ++f4) {
      float4 v = p[f4];
      r[4*f4+0]=v.x; r[4*f4+1]=v.y; r[4*f4+2]=v.z; r[4*f4+3]=v.w;
    }
#pragma unroll
    for (int d = 0; d < 8; ++d) {
      float acc = tval * Ws[d*65 + 64];
#pragma unroll
      for (int f = 0; f < 64; ++f) acc += r[f] * Ws[d*65 + f];
      q_ws[((size_t)bn * 8 + d) * T_LEN + t] = acc;
    }
  }
  { // K pass
    float r[64];
    const float4* p = (const float4*)(K + row);
#pragma unroll
    for (int f4 = 0; f4 < 16; ++f4) {
      float4 v = p[f4];
      r[4*f4+0]=v.x; r[4*f4+1]=v.y; r[4*f4+2]=v.z; r[4*f4+3]=v.w;
    }
#pragma unroll
    for (int d = 0; d < 8; ++d) {
      float acc = tval * Ws[d*65 + 64];
#pragma unroll
      for (int f = 0; f < 64; ++f) acc += r[f] * Ws[d*65 + f];
      k_ws[((size_t)bn * 8 + d) * T_LEN + t] = acc;
    }
  }
}

// ------- Kernel 2: circular correlation as bf16 WMMA GEMM (lag-tile trick) -------
// Per wave: C[M,N] = sum_sb sum_K q[tA+sb+K+M] * k[sb+K+16N] = corr[tA + M - 16N]
// tA = L0 + 240 -> one 16x16 tile covers lags [L0, L0+256).
__global__ __launch_bounds__(256) void corr_kernel(
    const float* __restrict__ q_ws, const float* __restrict__ k_ws,
    float* __restrict__ corr)
{
  __shared__ float q_s[4064];                         // q circularly extended (f32)
  __shared__ __align__(32) unsigned short k_s[2272];  // k circularly extended (bf16)
  int seq = blockIdx.x;
  const float* q = q_ws + (size_t)seq * T_LEN;
  const float* k = k_ws + (size_t)seq * T_LEN;

  for (int x = threadIdx.x; x < 4064; x += 256) {
    int t = x; if (t >= T_LEN) t -= T_LEN; if (t >= T_LEN) t -= T_LEN;
    q_s[x] = q[t];
  }
  for (int x = threadIdx.x; x < 2272; x += 256) {
    int t = x; if (t >= T_LEN) t -= T_LEN;
    __bf16 h = (__bf16)k[t];
    k_s[x] = __builtin_bit_cast(unsigned short, h);
  }
  __syncthreads();

  int lane = threadIdx.x & 31;
  int L0   = (threadIdx.x >> 5) << 8;  // lag-tile base per wave: 0,256,...,1792
  int i    = lane & 15;                // A row M / B column N
  int hi   = lane >> 4;
  int aoff = L0 + 240 + i + hi * 16;   // A: lanes16-31 carry K = 16..31
  int boff = (i << 4) + (hi << 4);     // B: 16*col + (hi?16:0) K-base

  v8f c = {};
  for (int sb = 0; sb < T_LEN; sb += 32) {
    v16bf a = {};
    int b0 = aoff + sb;
#pragma unroll
    for (int m = 0; m < 8; ++m) a[m]     = (__bf16)q_s[b0 + m];       // K = klo+0..7
#pragma unroll
    for (int m = 0; m < 8; ++m) a[8 + m] = (__bf16)q_s[b0 + 8 + m];   // K = klo+8..15
    v16bf bm = *(const v16bf*)(&k_s[sb + boff]);                      // 32B aligned
    c = __builtin_amdgcn_wmma_f32_16x16x32_bf16(false, a, false, bm,
                                                (short)0, c, false, false);
  }

  float* crow = corr + (size_t)seq * T_LEN;
#pragma unroll
  for (int r = 0; r < 8; ++r) {
    int M   = r + hi * 8;                 // C/D layout: VGPR r -> M=r / r+8
    int lag = L0 + M + ((15 - i) << 4);   // lag = tA + M - 16N
    if (lag < T_LEN) crow[lag] = c[r];
  }
}

// ---------------- Kernel 3: top-3 + softmax over 2016 lags ----------------
__global__ __launch_bounds__(256) void topk_kernel(
    const float* __restrict__ corr, float* __restrict__ Dout, float* __restrict__ Wout)
{
  int seq = blockIdx.x;
  const float* c = corr + (size_t)seq * T_LEN;
  float v0 = -1e30f, v1 = -1e30f, v2 = -1e30f;
  int   i0 = 0, i1 = 0, i2 = 0;
  for (int t = threadIdx.x; t < T_LEN; t += 256) {
    float x = c[t];
    if      (x > v0) { v2=v1; i2=i1; v1=v0; i1=i0; v0=x; i0=t; }
    else if (x > v1) { v2=v1; i2=i1; v1=x; i1=t; }
    else if (x > v2) { v2=x; i2=t; }
  }
  __shared__ float sv[768];
  __shared__ int   si[768];
  sv[threadIdx.x*3+0]=v0; si[threadIdx.x*3+0]=i0;
  sv[threadIdx.x*3+1]=v1; si[threadIdx.x*3+1]=i1;
  sv[threadIdx.x*3+2]=v2; si[threadIdx.x*3+2]=i2;
  __syncthreads();
  if (threadIdx.x == 0) {
    float b0=-1e30f,b1=-1e30f,b2=-1e30f; int j0=0,j1=0,j2=0;
    for (int n = 0; n < 768; ++n) {
      float x = sv[n]; int ix = si[n];
      if      (x > b0) { b2=b1; j2=j1; b1=b0; j1=j0; b0=x; j0=ix; }
      else if (x > b1) { b2=b1; j2=j1; b1=x; j1=ix; }
      else if (x > b2) { b2=x; j2=ix; }
    }
    float e0 = __expf(b0 - b0), e1 = __expf(b1 - b0), e2 = __expf(b2 - b0);
    float inv = 1.0f / (e0 + e1 + e2);
    Dout[seq*3+0] = (float)j0; Wout[seq*3+0] = e0 * inv;
    Dout[seq*3+1] = (float)j1; Wout[seq*3+1] = e1 * inv;
    Dout[seq*3+2] = (float)j2; Wout[seq*3+2] = e2 * inv;
  }
}

// ---------------- Kernel 4: delay-weighted aggregation ----------------
__global__ __launch_bounds__(256) void agg_kernel(
    const float* __restrict__ V, const float* __restrict__ Dsec,
    const float* __restrict__ Wsec, float* __restrict__ out)
{
  int bn = blockIdx.x;
  __shared__ float wsh[24];
  __shared__ int   dsh[24];
  if (threadIdx.x < 24) {
    wsh[threadIdx.x] = Wsec[bn*24 + threadIdx.x] * 0.125f;  // fold 1/H
    dsh[threadIdx.x] = (int)Dsec[bn*24 + threadIdx.x];
  }
  __syncthreads();
  for (int t = threadIdx.x; t < T_LEN; t += 256) {
    float a0=0,a1=0,a2=0,a3=0,a4=0,a5=0,a6=0,a7=0;
#pragma unroll
    for (int p = 0; p < 24; ++p) {
      int tp = t + dsh[p]; if (tp >= T_LEN) tp -= T_LEN;   // circular (vt=[v,v])
      const float4* vr = (const float4*)(V + ((size_t)bn * T_LEN + tp) * 64);
      float4 x0 = vr[0], x1 = vr[1];
      float wp = wsh[p];
      a0 += wp*x0.x; a1 += wp*x0.y; a2 += wp*x0.z; a3 += wp*x0.w;
      a4 += wp*x1.x; a5 += wp*x1.y; a6 += wp*x1.z; a7 += wp*x1.w;
    }
    float4* op = (float4*)(out + ((size_t)bn * T_LEN + t) * 8);
    float4 o0 = {a0,a1,a2,a3}, o1 = {a4,a5,a6,a7};
    op[0] = o0; op[1] = o1;
  }
}

extern "C" void kernel_launch(void* const* d_in, const int* in_sizes, int n_in,
                              void* d_out, int out_size, void* d_ws, size_t ws_size,
                              hipStream_t stream) {
  const float* Q  = (const float*)d_in[0];
  const float* K  = (const float*)d_in[1];
  const float* V  = (const float*)d_in[2];
  const float* tt = (const float*)d_in[3];
  const float* W  = (const float*)d_in[4];

  float* out  = (float*)d_out;                        // (2,128,2016,8)
  float* Dout = out + (size_t)NBN * T_LEN * 8;        // (2,128,8,3) as float
  float* Wout = Dout + (size_t)NSEQ * 3;              // (2,128,8,3)

  float* ws   = (float*)d_ws;
  float* q_ws = ws;                                   // NSEQ*T f32
  float* k_ws = q_ws + (size_t)NSEQ * T_LEN;          // NSEQ*T f32
  float* corr = k_ws + (size_t)NSEQ * T_LEN;          // NSEQ*T f32

  proj_kernel<<<NBN * 8, 256, 0, stream>>>(Q, K, tt, W, q_ws, k_ws);
  corr_kernel<<<NSEQ,    256, 0, stream>>>(q_ws, k_ws, corr);
  topk_kernel<<<NSEQ,    256, 0, stream>>>(corr, Dout, Wout);
  agg_kernel <<<NBN,     256, 0, stream>>>(V, Dout, Wout, out);
}
